// MultiHeadAttention_1236950581691
// MI455X (gfx1250) — matrix operand order repaired and checked by reading, not compiled, not run on hardware
//
#include <hip/hip_runtime.h>
#include <hip/hip_bf16.h>
#include <stdint.h>

// ---------------------------------------------------------------------------
// MHA + LayerNorm for MI455X (gfx1250, wave32, WMMA f32_16x16x32_f16)
//   D_MODEL=1024, D_K=D_V=64, H=16, B=2, S=2048
// Pipeline:
//   0) weight f32->f16 transpose (LDS-tiled)           [run 4x: Wq,Wk,Wv,Wo]
//   1) QKV projection GEMM (templated output layout; no runtime branching)
//   2) flash attention (online softmax, no global score traffic)
//   3) output projection + residual + LayerNorm (ds_add_f32 row stats)
// All fragment loads are aligned 16B/32B vector loads; all index math 32-bit.
// ---------------------------------------------------------------------------

typedef __attribute__((ext_vector_type(16))) _Float16 v16h;
typedef __attribute__((ext_vector_type(8)))  _Float16 v8h;
typedef __attribute__((ext_vector_type(8)))  float    v8f;

#define D_MODEL 1024
#define D_HEAD  64
#define N_HEADS 16
#define BATCH   2
#define SEQ     2048
#define ROWS    (BATCH * SEQ)      // 4096
#define LN_EPS  1e-5f

// ---------------------------------------------------------------------------
// Fragment loaders (contiguous k order k = 16*(lane>>4) + s in A and B, wave32)
// ---------------------------------------------------------------------------

// A fragment 16x32 f16 from row-major f16 (stride lda halves).
// lane L: row = L&15; halves h -> K = (L>>4)*16 + h. Two aligned 16B loads.
static __device__ __forceinline__ v16h load_a_f16(const _Float16* __restrict__ src, int lda) {
  const int lane = threadIdx.x & 31;
  const _Float16* p = src + (lane & 15) * lda + ((lane >> 4) << 4);
  v8h lo = *(const v8h*)p;
  v8h hi = *(const v8h*)(p + 8);
  return __builtin_shufflevector(lo, hi, 0, 1, 2, 3, 4, 5, 6, 7,
                                 8, 9, 10, 11, 12, 13, 14, 15);
}

// A fragment from row-major f32 with conversion. Two aligned 32B loads + cvt.
static __device__ __forceinline__ v16h load_a_f32cvt(const float* __restrict__ src, int lda) {
  const int lane = threadIdx.x & 31;
  const float* p = src + (lane & 15) * lda + ((lane >> 4) << 4);
  v8h lo = __builtin_convertvector(*(const v8f*)p, v8h);
  v8h hi = __builtin_convertvector(*(const v8f*)(p + 8), v8h);
  return __builtin_shufflevector(lo, hi, 0, 1, 2, 3, 4, 5, 6, 7,
                                 8, 9, 10, 11, 12, 13, 14, 15);
}

// B fragment 32x16 f16, K-CONTIGUOUS source: element (k,n) at src[n*ldn + k].
// lane: col n = lane&15, halves h -> K = (lane>>4)*16 + h. One aligned 32B load.
static __device__ __forceinline__ v16h load_b_ctg(const _Float16* __restrict__ src, int ldn) {
  const int lane = threadIdx.x & 31;
  return *(const v16h*)(src + (lane & 15) * ldn + ((lane >> 4) << 4));
}

static __device__ __forceinline__ v8f wmma_f16(v16h a, v16h b, v8f c) {
  return __builtin_amdgcn_wmma_f32_16x16x32_f16(false, a, false, b, (short)0, c, false, false);
}

// ---------------------------------------------------------------------------
// Kernel 0: Wt[n,k] = (f16) W[k,n]   (1024x1024), LDS-tiled transpose.
// ---------------------------------------------------------------------------
__global__ __launch_bounds__(256) void wt_cvt_kernel(
    const float* __restrict__ W, _Float16* __restrict__ Wt) {
  __shared__ _Float16 tile[32][33];
  const int tx = threadIdx.x & 31;
  const int ty = threadIdx.x >> 5;          // 0..7
  const int k0 = (blockIdx.x & 31) * 32;
  const int n0 = (blockIdx.x >> 5) * 32;
#pragma unroll
  for (int r = 0; r < 32; r += 8)
    tile[ty + r][tx] = (_Float16)W[(k0 + ty + r) * D_MODEL + n0 + tx];
  __syncthreads();
#pragma unroll
  for (int r = 0; r < 32; r += 8)
    Wt[(n0 + ty + r) * D_MODEL + k0 + tx] = tile[tx][ty + r];
}

// ---------------------------------------------------------------------------
// Kernel 1: Y = X * W + bias. Wt is f16 [n,k]. Per wave: 16x64 output tile.
// TRANSPOSED=0: store [B,H,S,64] (q,k); TRANSPOSED=1: store [B,H,64,S] (v).
// ---------------------------------------------------------------------------
template <int TRANSPOSED>
__global__ __launch_bounds__(256) void qkv_proj_kernel(
    const float* __restrict__ X, const _Float16* __restrict__ Wt,
    const float* __restrict__ bias, _Float16* __restrict__ Yhat) {
  const int wave = threadIdx.x >> 5;
  const int lane = threadIdx.x & 31;
  const int tile = blockIdx.x * 8 + wave;
  const int r0   = (tile >> 4) * 16;
  const int n0   = (tile & 15) * 64;

  v8f z = {};
  v8f acc[4] = {z, z, z, z};

  const float*    xp = X + r0 * D_MODEL;
  const _Float16* wp = Wt + n0 * D_MODEL;
  for (int k = 0; k < D_MODEL; k += 32) {
    v16h a = load_a_f32cvt(xp + k, D_MODEL);
#pragma unroll
    for (int j = 0; j < 4; ++j)
      acc[j] = wmma_f16(a, load_b_ctg(wp + j * 16 * D_MODEL + k, D_MODEL), acc[j]);
  }

  const int nl = lane & 15;
  const int mh = (lane >> 4) << 3;
#pragma unroll
  for (int j = 0; j < 4; ++j) {
    const int c  = n0 + j * 16 + nl;
    const int hh = c >> 6;
    const int d  = c & 63;
    const float bv = bias[c];
#pragma unroll
    for (int i = 0; i < 8; ++i) {
      const int r  = r0 + mh + i;
      const int bb = r >> 11;              // r / SEQ
      const int s  = r & (SEQ - 1);
      int idx;
      if (TRANSPOSED)
        idx = ((bb * N_HEADS + hh) * D_HEAD + d) * SEQ + s;    // [B,H,D,S]
      else
        idx = ((bb * N_HEADS + hh) * SEQ + s) * D_HEAD + d;    // [B,H,S,D]
      Yhat[idx] = (_Float16)(acc[j][i] + bv);
    }
  }
}

// ---------------------------------------------------------------------------
// Kernel 2: flash attention. Block = (b, h, 128 q-rows); wave = 16 q-rows.
// kh: [B,H,S,64] (d-contiguous B frags); vt: [B,H,64,S] (key-contiguous).
// ---------------------------------------------------------------------------
__global__ __launch_bounds__(256) void flash_attn_kernel(
    const _Float16* __restrict__ qh, const _Float16* __restrict__ kh,
    const _Float16* __restrict__ vt, const unsigned char* __restrict__ mask,
    _Float16* __restrict__ ctx) {
  __shared__ _Float16 plds[8][16 * 64];   // per-wave 16x64 P tile (16 KB)

  const int wave = threadIdx.x >> 5;
  const int lane = threadIdx.x & 31;
  const int blk  = blockIdx.x;            // b*256 + h*16 + qb
  const int qb   = blk & 15;
  const int h    = (blk >> 4) & 15;
  const int b    = blk >> 8;
  const int base = (b * N_HEADS + h) * SEQ * D_HEAD;  // < 2^23, fits int
  const int q0   = qb * 128 + wave * 16;

  const _Float16* kbase = kh + base;
  const _Float16* vbase = vt + base;

  const v16h aq0 = load_a_f16(qh + base + q0 * D_HEAD, D_HEAD);
  const v16h aq1 = load_a_f16(qh + base + q0 * D_HEAD + 32, D_HEAD);

  v8f z = {};
  v8f o[4] = {z, z, z, z};
  float mrun[8], lrun[8];
#pragma unroll
  for (int i = 0; i < 8; ++i) { mrun[i] = -3.0e38f; lrun[i] = 0.0f; }

  const int nl = lane & 15;
  const int mh = (lane >> 4) << 3;
  // mask row pointer for this lane's first row; row stride = SEQ bytes
  const unsigned char* mrow0 = mask + (b * SEQ + q0 + mh) * SEQ + nl;

  for (int kb = 0; kb < SEQ; kb += 64) {
    // ---- S = Q K^T over this 64-key block (8 WMMAs) ----
    v8f sc[4] = {z, z, z, z};
#pragma unroll
    for (int j = 0; j < 4; ++j) {
      const _Float16* kp = kbase + (kb + j * 16) * D_HEAD;
      sc[j] = wmma_f16(aq0, load_b_ctg(kp, D_HEAD), sc[j]);
      sc[j] = wmma_f16(aq1, load_b_ctg(kp + 32, D_HEAD), sc[j]);
    }

    // ---- scale + mask ----
#pragma unroll
    for (int i = 0; i < 8; ++i) {
      const unsigned char* mp = mrow0 + i * SEQ + kb;
#pragma unroll
      for (int j = 0; j < 4; ++j) {
        float s = sc[j][i] * 0.125f;              // 1/sqrt(64)
        if (mp[j * 16]) s = -1.0e9f;              // mask==true -> -1e9
        sc[j][i] = s;
      }
    }

    // ---- online softmax (row = 16-lane half at each VGPR index) ----
#pragma unroll
    for (int i = 0; i < 8; ++i) {
      float rm = fmaxf(fmaxf(sc[0][i], sc[1][i]), fmaxf(sc[2][i], sc[3][i]));
      rm = fmaxf(rm, __shfl_xor(rm, 1, 16));
      rm = fmaxf(rm, __shfl_xor(rm, 2, 16));
      rm = fmaxf(rm, __shfl_xor(rm, 4, 16));
      rm = fmaxf(rm, __shfl_xor(rm, 8, 16));
      const float newm = fmaxf(mrun[i], rm);
      const float corr = __expf(mrun[i] - newm);
      mrun[i] = newm;

      float rs = 0.0f;
#pragma unroll
      for (int j = 0; j < 4; ++j) {
        const float p = __expf(sc[j][i] - newm);
        sc[j][i] = p;
        rs += p;
      }
      rs += __shfl_xor(rs, 1, 16);
      rs += __shfl_xor(rs, 2, 16);
      rs += __shfl_xor(rs, 4, 16);
      rs += __shfl_xor(rs, 8, 16);
      lrun[i] = lrun[i] * corr + rs;

#pragma unroll
      for (int j = 0; j < 4; ++j) o[j][i] *= corr;

      // C-layout -> LDS f16 (re-read below in A-layout; same-wave DS ordered)
#pragma unroll
      for (int j = 0; j < 4; ++j)
        plds[wave][(mh + i) * 64 + j * 16 + nl] = (_Float16)sc[j][i];
    }

    // ---- O += P V (8 WMMAs); V B-frags are key-contiguous 32B loads ----
#pragma unroll
    for (int ks = 0; ks < 2; ++ks) {
      v16h pa = load_a_f16(&plds[wave][ks * 32], 64);
#pragma unroll
      for (int j = 0; j < 4; ++j) {
        // element(k=key_rel, n=d): vt[base + (j*16+n)*SEQ + kb + ks*32 + k]
        v16h bv = load_b_ctg(vbase + (j * 16) * SEQ + kb + ks * 32, SEQ);
        o[j] = wmma_f16(pa, bv, o[j]);
      }
    }
  }

  // ---- epilogue: normalize, store context f16 in [B, S, H*64] ----
#pragma unroll
  for (int i = 0; i < 8; ++i) {
    const float inv = 1.0f / lrun[i];
    const int qrow = q0 + mh + i;
#pragma unroll
    for (int j = 0; j < 4; ++j) {
      ctx[(b * SEQ + qrow) * (N_HEADS * D_HEAD) + h * D_HEAD + j * 16 + nl] =
          (_Float16)(o[j][i] * inv);
    }
  }
}

// ---------------------------------------------------------------------------
// Kernel 3: out = LN(ctx * Wo + bo + Q). Block = 16 rows x 1024 cols.
// Wo_t is f16 [n,k]. Cross-wave row stats via ds_add_f32.
// ---------------------------------------------------------------------------
__global__ __launch_bounds__(256) void outproj_ln_kernel(
    const _Float16* __restrict__ ctx, const _Float16* __restrict__ Wot,
    const float* __restrict__ bo, const float* __restrict__ Qin,
    const float* __restrict__ gamma, const float* __restrict__ beta,
    float* __restrict__ out) {
  __shared__ float ssum[16];
  __shared__ float ssq[16];

  const int wave = threadIdx.x >> 5;
  const int lane = threadIdx.x & 31;
  const int r0 = blockIdx.x * 16;
  const int n0 = wave * 128;

  if (threadIdx.x < 16)       ssum[threadIdx.x] = 0.0f;
  else if (threadIdx.x < 32)  ssq[threadIdx.x - 16] = 0.0f;
  __syncthreads();

  v8f z = {};
  v8f acc[8] = {z, z, z, z, z, z, z, z};

  const _Float16* cp = ctx + r0 * D_MODEL;
  const _Float16* wp = Wot + n0 * D_MODEL;
  for (int k = 0; k < D_MODEL; k += 32) {
    v16h a = load_a_f16(cp + k, D_MODEL);
#pragma unroll
    for (int j = 0; j < 8; ++j)
      acc[j] = wmma_f16(a, load_b_ctg(wp + j * 16 * D_MODEL + k, D_MODEL), acc[j]);
  }

  const int nl = lane & 15;
  const int mh = (lane >> 4) << 3;

#pragma unroll
  for (int i = 0; i < 8; ++i) {
    const int r = r0 + mh + i;
    float ps = 0.0f, ps2 = 0.0f;
#pragma unroll
    for (int j = 0; j < 8; ++j) {
      const int c = n0 + j * 16 + nl;
      const float v = acc[j][i] + bo[c] + Qin[r * D_MODEL + c];  // + residual
      acc[j][i] = v;
      ps  += v;
      ps2 += v * v;
    }
    ps  += __shfl_xor(ps, 1, 16);  ps  += __shfl_xor(ps, 2, 16);
    ps  += __shfl_xor(ps, 4, 16);  ps  += __shfl_xor(ps, 8, 16);
    ps2 += __shfl_xor(ps2, 1, 16); ps2 += __shfl_xor(ps2, 2, 16);
    ps2 += __shfl_xor(ps2, 4, 16); ps2 += __shfl_xor(ps2, 8, 16);
    if (nl == 0) {
      atomicAdd(&ssum[mh + i], ps);
      atomicAdd(&ssq[mh + i], ps2);
    }
  }
  __syncthreads();

#pragma unroll
  for (int i = 0; i < 8; ++i) {
    const int r = r0 + mh + i;
    const float mu   = ssum[mh + i] * (1.0f / D_MODEL);
    const float var  = ssq[mh + i] * (1.0f / D_MODEL) - mu * mu;
    const float rstd = rsqrtf(var + LN_EPS);
#pragma unroll
    for (int j = 0; j < 8; ++j) {
      const int c = n0 + j * 16 + nl;
      out[r * D_MODEL + c] = (acc[j][i] - mu) * rstd * gamma[c] + beta[c];
    }
  }
}

// ---------------------------------------------------------------------------
// Launch
// ---------------------------------------------------------------------------
extern "C" void kernel_launch(void* const* d_in, const int* in_sizes, int n_in,
                              void* d_out, int out_size, void* d_ws, size_t ws_size,
                              hipStream_t stream) {
  (void)in_sizes; (void)n_in; (void)out_size; (void)ws_size;

  const float* Q    = (const float*)d_in[0];
  const float* K    = (const float*)d_in[1];
  const float* V    = (const float*)d_in[2];
  const unsigned char* mask = (const unsigned char*)d_in[3];  // jnp bool_ = 1 byte
  const float* Wq   = (const float*)d_in[4];
  const float* bq   = (const float*)d_in[5];
  const float* Wk   = (const float*)d_in[6];
  const float* bk   = (const float*)d_in[7];
  const float* Wv   = (const float*)d_in[8];
  const float* bv   = (const float*)d_in[9];
  const float* Wo   = (const float*)d_in[10];
  const float* bo   = (const float*)d_in[11];
  const float* gamma= (const float*)d_in[12];
  const float* beta = (const float*)d_in[13];
  float* outp = (float*)d_out;

  // workspace (f16): qh|kh|vt|ctx (4 x 4M halves) + 4 weight transposes (1M each)
  const size_t NELEM = (size_t)ROWS * D_MODEL;    // 4 Mi
  const size_t WELEM = (size_t)D_MODEL * D_MODEL; // 1 Mi
  _Float16* qh  = (_Float16*)d_ws;
  _Float16* kh  = qh  + NELEM;
  _Float16* vt  = kh  + NELEM;
  _Float16* ctx = vt  + NELEM;
  _Float16* wqt = ctx + NELEM;
  _Float16* wkt = wqt + WELEM;
  _Float16* wvt = wkt + WELEM;
  _Float16* wot = wvt + WELEM;

  wt_cvt_kernel<<<1024, 256, 0, stream>>>(Wq, wqt);
  wt_cvt_kernel<<<1024, 256, 0, stream>>>(Wk, wkt);
  wt_cvt_kernel<<<1024, 256, 0, stream>>>(Wv, wvt);
  wt_cvt_kernel<<<1024, 256, 0, stream>>>(Wo, wot);

  qkv_proj_kernel<0><<<512, 256, 0, stream>>>(Q, wqt, bq, qh);
  qkv_proj_kernel<0><<<512, 256, 0, stream>>>(K, wkt, bk, kh);
  qkv_proj_kernel<1><<<512, 256, 0, stream>>>(V, wvt, bv, vt);

  flash_attn_kernel<<<512, 256, 0, stream>>>(qh, kh, vt, mask, ctx);
  outproj_ln_kernel<<<256, 256, 0, stream>>>(ctx, wot, bo, Q, gamma, beta, outp);
}
